// CSWinBlock_8375186227704
// MI455X (gfx1250) — matrix operand order repaired and checked by reading, not compiled, not run on hardware
//
#include <hip/hip_runtime.h>
#include <hip/hip_bf16.h>
#include <math.h>

// ---------------------------------------------------------------------------
// CDNA5 (gfx1250, wave32) CSWin block. All GEMM-shaped math goes through
// v_wmma_f32_16x16x32_bf16. B operands are pre-transposed (N x K, bf16) so
// each lane's B fragment is two 8-half K runs, 16 apart (2x ds/global b128 loads).
// GEMM staging uses GLOBAL_LOAD_ASYNC_TO_LDS_B128 (ASYNCcnt) with LDS
// double-buffering when the toolchain exposes the gfx1250 async builtins.
// ---------------------------------------------------------------------------

typedef __bf16 bf16;
typedef __attribute__((ext_vector_type(16))) __bf16 v16bf;
typedef __attribute__((ext_vector_type(8)))  float  v8f;
typedef int v4i_vec __attribute__((vector_size(16)));   // matches builtin param

union FragAB {
    v16bf v;
    uint4 q[2];
};

__device__ __forceinline__ v8f wmma_bb(v16bf a, v16bf b, v8f c) {
    // 8 args: (neg_a, A, neg_b, B, c_mod, C, reuse_a, reuse_b)
    return __builtin_amdgcn_wmma_f32_16x16x32_bf16(false, a, false, b,
                                                   (short)0, c, false, false);
}

__device__ __forceinline__ v8f v8f_zero() {
    v8f z = {0.f, 0.f, 0.f, 0.f, 0.f, 0.f, 0.f, 0.f};
    return z;
}

// Async global -> LDS 16B copy (falls back to sync VGPR bounce if the
// builtin is not declared by this toolchain).
#if defined(__AMDGCN__) && __has_builtin(__builtin_amdgcn_global_load_async_to_lds_b128)
#define HAVE_ASYNC_LDS 1
#else
#define HAVE_ASYNC_LDS 0
#endif

__device__ __forceinline__ void stage16(bf16* lds_dst, const bf16* gsrc) {
#if HAVE_ASYNC_LDS
    __builtin_amdgcn_global_load_async_to_lds_b128(
        (__attribute__((address_space(1))) v4i_vec*)gsrc,
        (__attribute__((address_space(3))) v4i_vec*)lds_dst, 0, 0);
#else
    *(uint4*)lds_dst = *(const uint4*)gsrc;
#endif
}

__device__ __forceinline__ void wait_async_copies() {
#if HAVE_ASYNC_LDS && __has_builtin(__builtin_amdgcn_s_wait_asynccnt)
    __builtin_amdgcn_s_wait_asynccnt(0);
#endif
}

// ---------------------------------------------------------------------------
// Weight transpose + f32 -> bf16 conversion:  W (K x N, f32) -> Bt (N x K, bf16)
// ---------------------------------------------------------------------------
__global__ void transpose_to_bt(const float* __restrict__ W, bf16* __restrict__ Bt,
                                int K, int N) {
    long idx = (long)blockIdx.x * blockDim.x + threadIdx.x;
    if (idx >= (long)K * N) return;
    int n = (int)(idx / K);
    int k = (int)(idx - (long)n * K);
    Bt[idx] = (bf16)W[(long)k * N + n];
}

// sg_dil_w (Co=512, Ci=512, 3, 3) OIHW -> 9 tap matrices Bt_t[co][ci] (bf16)
__global__ void dilw_to_bt(const float* __restrict__ W, bf16* __restrict__ Bt) {
    long idx = (long)blockIdx.x * blockDim.x + threadIdx.x;
    if (idx >= (long)9 * 512 * 512) return;
    int t  = (int)(idx / (512 * 512));
    int r  = (int)(idx - (long)t * 512 * 512);
    int co = r / 512;
    int ci = r - co * 512;
    Bt[(long)t * 512 * 512 + (long)co * 512 + ci] =
        (bf16)W[((long)co * 512 + ci) * 9 + t];
}

// ---------------------------------------------------------------------------
// LayerNorm over C channels -> bf16.  One block per token, 256 threads.
// Input is f32 (inf) or bf16 (inb); stride/off select a channel slice.
// ---------------------------------------------------------------------------
__global__ __launch_bounds__(256)
void ln_kernel(const float* __restrict__ inf, const bf16* __restrict__ inb,
               const float* __restrict__ g, const float* __restrict__ bsh,
               bf16* __restrict__ out, int C, int stride, int off) {
    long t = blockIdx.x;
    float vals[2];
    int nv = 0;
    float s = 0.f, s2 = 0.f;
    for (int c = threadIdx.x; c < C; c += 256) {
        long src = t * (long)stride + off + c;
        float v = inb ? (float)inb[src] : inf[src];
        vals[nv++] = v;
        s += v; s2 += v * v;
    }
    __shared__ float rs[8], rs2[8];
    for (int o = 16; o > 0; o >>= 1) {
        s  += __shfl_xor(s, o, 32);
        s2 += __shfl_xor(s2, o, 32);
    }
    int wave = threadIdx.x >> 5, lane = threadIdx.x & 31;
    if (lane == 0) { rs[wave] = s; rs2[wave] = s2; }
    __syncthreads();
    if (wave == 0) {
        s  = (lane < 8) ? rs[lane]  : 0.f;
        s2 = (lane < 8) ? rs2[lane] : 0.f;
        for (int o = 4; o > 0; o >>= 1) {
            s  += __shfl_xor(s, o, 32);
            s2 += __shfl_xor(s2, o, 32);
        }
        if (lane == 0) { rs[0] = s; rs2[0] = s2; }
    }
    __syncthreads();
    float mean = rs[0] / (float)C;
    float var  = rs2[0] / (float)C - mean * mean;
    float rstd = rsqrtf(var + 1e-5f);
    nv = 0;
    for (int c = threadIdx.x; c < C; c += 256) {
        out[t * (long)C + c] = (bf16)((vals[nv++] - mean) * rstd * g[c] + bsh[c]);
    }
}

// ---------------------------------------------------------------------------
// Generic WMMA GEMM:  C(M x N) = A(M x K, bf16 row-major) @ Bt(N x K, bf16)^T
// Block tile 128x128, BK=32, 256 threads = 8 waves in a 2x4 wave grid,
// each wave owns 4x2 fragments of 16x16 (f32 acc).  LDS is double buffered;
// tile t+1 is staged (async if available) while tile t feeds the WMMA pipe.
// mode 0: bf16 out (+bias);  1: bf16 gelu(acc+bias);
// mode 2: f32 out = acc + bias + resid;  3: f32 accumulate (acc_load).
// dh/dw != 0: A rows gathered with 2D shift + zero pad (implicit dilated conv).
// ---------------------------------------------------------------------------
struct GemmParams {
    const bf16* A; const bf16* Bt;
    float* Cf; bf16* Cb;
    const float* bias; const float* resid;
    int M, N, K;
    int mode, acc_load;
    int dh, dw, Hd, Wd;
};

__global__ __launch_bounds__(256)
void gemm_wmma(GemmParams p) {
    __shared__ __align__(16) bf16 As[2][128 * 40];   // padded: 32 halves + 8
    __shared__ __align__(16) bf16 Bs[2][128 * 40];
    const int tid = threadIdx.x;
    const int wave = tid >> 5, lane = tid & 31;
    const int lq = lane & 15, lh = lane >> 4;
    const int wm = wave >> 2, wn = wave & 3;
    const long m0 = (long)blockIdx.y * 128;
    const long n0 = (long)blockIdx.x * 128;
    const int sr = tid & 127;     // staging row within tile
    const int sc = tid >> 7;      // staging chunk (16 halves each)

    long arow; bool avalid = true;
    {
        long gm = m0 + sr;
        if (p.dh | p.dw) {
            int L = p.Hd * p.Wd;
            long bb = gm / L;
            int pos = (int)(gm - bb * L);
            int h = pos / p.Wd + p.dh;
            int w = pos - (pos / p.Wd) * p.Wd + p.dw;
            avalid = (h >= 0) && (h < p.Hd) && (w >= 0) && (w < p.Wd);
            arow = bb * L + (long)h * p.Wd + w;
        } else {
            arow = gm;
        }
    }

    v8f acc[4][2];
#pragma unroll
    for (int i = 0; i < 4; ++i)
#pragma unroll
        for (int j = 0; j < 2; ++j) acc[i][j] = v8f_zero();

    auto stage = [&](int buf, int k0) {
        bf16* ad = &As[buf][sr * 40 + sc * 16];
        if (avalid) {
            const bf16* as = p.A + arow * p.K + k0 + sc * 16;
            __builtin_prefetch(p.A + arow * p.K + k0 + 64, 0, 1);
            stage16(ad, as);
            stage16(ad + 8, as + 8);
        } else {
            const uint4 z = {0u, 0u, 0u, 0u};
            *(uint4*)ad = z;
            *(uint4*)(ad + 8) = z;
        }
        bf16* bd = &Bs[buf][sr * 40 + sc * 16];
        const bf16* bsrc = p.Bt + (n0 + sr) * p.K + k0 + sc * 16;
        stage16(bd, bsrc);
        stage16(bd + 8, bsrc + 8);
    };

    const int nT = p.K >> 5;
    stage(0, 0);
    for (int t = 0; t < nT; ++t) {
        wait_async_copies();      // own async writes to LDS landed
        __syncthreads();          // everyone's tile-t data visible; tile t-1 reads done
        if (t + 1 < nT) stage((t + 1) & 1, (t + 1) * 32);

        const bf16* Ab = As[t & 1];
        const bf16* Bb = Bs[t & 1];
        FragAB afr[4], bfr[2];
#pragma unroll
        for (int fn = 0; fn < 2; ++fn) {
            // B frag: lane holds K = lh*8..+7 and 16+lh*8..+7 for column n
            const bf16* bp = Bb + (wn * 32 + fn * 16 + lq) * 40 + lh * 8;
            bfr[fn].q[0] = *(const uint4*)bp;
            bfr[fn].q[1] = *(const uint4*)(bp + 16);
        }
#pragma unroll
        for (int fm = 0; fm < 4; ++fm) {
            // A frag: lane holds K = lh*8..+7 and 16+lh*8..+7 for row m
            const bf16* ap = Ab + (wm * 64 + fm * 16 + lq) * 40;
            afr[fm].q[0] = *(const uint4*)(ap + lh * 8);
            afr[fm].q[1] = *(const uint4*)(ap + 16 + lh * 8);
        }
#pragma unroll
        for (int fm = 0; fm < 4; ++fm)
#pragma unroll
            for (int fn = 0; fn < 2; ++fn)
                acc[fm][fn] = wmma_bb(afr[fm].v, bfr[fn].v, acc[fm][fn]);
        // no tail barrier: next iteration's top barrier provides the ordering
    }

    // epilogue: C layout -> lane lq = col, VGPR r -> row r + 8*lh
#pragma unroll
    for (int fm = 0; fm < 4; ++fm) {
#pragma unroll
        for (int fn = 0; fn < 2; ++fn) {
#pragma unroll
            for (int r = 0; r < 8; ++r) {
                long grow = m0 + wm * 64 + fm * 16 + r + 8 * lh;
                long gcol = n0 + wn * 32 + fn * 16 + lq;
                long idx = grow * p.N + gcol;
                float v = acc[fm][fn][r];
                if (p.mode == 0) {
                    if (p.bias) v += p.bias[gcol];
                    p.Cb[idx] = (bf16)v;
                } else if (p.mode == 1) {
                    v += p.bias[gcol];
                    v = 0.5f * v * (1.0f + erff(v * 0.70710678118f));
                    p.Cb[idx] = (bf16)v;
                } else if (p.mode == 2) {
                    v += p.bias[gcol] + p.resid[idx];
                    p.Cf[idx] = v;
                } else {
                    if (p.acc_load) v += p.Cf[idx];
                    p.Cf[idx] = v;
                }
            }
        }
    }
}

// ---------------------------------------------------------------------------
// Cross-shaped window attention branch (WMMA).  One workgroup (4 waves) per
// (window, head, 32-row q tile).  Full scores (32 x Nw, f32) + V^T staged in
// LDS (fits: Nw=1024 -> 198 KB < 320 KB/WGP).  qkv rows: [q(256) k(256) v(256)].
// ---------------------------------------------------------------------------
struct AttnParams {
    const bf16* qkv; bf16* att;
    int cq0, Hsp, Wsp, nHw, nWw;
};

__global__ __launch_bounds__(128)
void attn_kernel(AttnParams p) {
    extern __shared__ __align__(16) char smem_attn[];
    const int Nw   = p.Hsp * p.Wsp;
    const int srow = Nw + 8;                 // f32 elements per score row
    const int vrow = Nw + 8;                 // halves per Vt row
    float* S  = (float*)smem_attn;                          // 32 x srow (f32)
    bf16*  Vt = (bf16*)(smem_attn + (size_t)32 * srow * 4); // 32 x vrow (bf16)

    const int nW  = p.nHw * p.nWw;
    const int bw  = blockIdx.y;
    const int b   = bw / nW;
    const int wi  = bw - b * nW;
    const int whi = wi / p.nWw, wwi = wi - (wi / p.nWw) * p.nWw;
    const int head = blockIdx.z;
    const int qt   = blockIdx.x;
    const int tid = threadIdx.x, wave = tid >> 5, lane = tid & 31;
    const int lq = lane & 15, lh = lane >> 4;
    const int qch = p.cq0 + head * 32;
    const float scale = 0.17677669529663687f;   // 32^-0.5

    auto tokrow = [&](int r) -> long {
        int h = whi * p.Hsp + r / p.Wsp;
        int w = wwi * p.Wsp + (r - (r / p.Wsp) * p.Wsp);
        return (long)b * 1024 + h * 32 + w;
    };

    // Q fragments (A layout), loaded straight from global
    FragAB qf[2];
#pragma unroll
    for (int f = 0; f < 2; ++f) {
        long t = tokrow(qt * 32 + f * 16 + lq);
        const bf16* src = p.qkv + t * 768 + qch;
        qf[f].q[0] = *(const uint4*)(src + lh * 8);
        qf[f].q[1] = *(const uint4*)(src + 16 + lh * 8);
    }
    // Stage V^T (transpose during LDS fill)
    for (int r = tid; r < Nw; r += 128) {
        long t = tokrow(r);
        const bf16* src = p.qkv + t * 768 + 512 + qch;
#pragma unroll
        for (int d = 0; d < 32; ++d) Vt[d * vrow + r] = src[d];
    }

    // S = scale * Q @ K^T  (K rows are natural Bt layout: hd contiguous)
    for (int kb = wave * 16; kb < Nw; kb += 64) {
        FragAB kf;
        {
            long t = tokrow(kb + lq);
            const bf16* src = p.qkv + t * 768 + 256 + qch;
            kf.q[0] = *(const uint4*)(src + lh * 8);
            kf.q[1] = *(const uint4*)(src + lh * 8 + 16);
        }
#pragma unroll
        for (int f = 0; f < 2; ++f) {
            v8f s = v8f_zero();
            s = wmma_bb(qf[f].v, kf.v, s);
#pragma unroll
            for (int r = 0; r < 8; ++r) {
                int row = f * 16 + r + 8 * lh;
                S[(size_t)row * srow + kb + lq] = s[r] * scale;
            }
        }
    }
    __syncthreads();

    // Row softmax; P (bf16) written aliased over the low half of each S row.
    for (int rr = 0; rr < 8; ++rr) {
        int row = wave * 8 + rr;
        float* Sr = S + (size_t)row * srow;
        float mx = -3.4e38f;
        for (int k = lane; k < Nw; k += 32) mx = fmaxf(mx, Sr[k]);
        for (int o = 16; o > 0; o >>= 1) mx = fmaxf(mx, __shfl_xor(mx, o, 32));
        float sum = 0.f;
        for (int k = lane; k < Nw; k += 32) {
            float e = expf(Sr[k] - mx);
            Sr[k] = e;
            sum += e;
        }
        for (int o = 16; o > 0; o >>= 1) sum += __shfl_xor(sum, o, 32);
        float inv = 1.f / sum;
        bf16* Pr = (bf16*)Sr;
        for (int k = lane; k < Nw; k += 32) {
            float pv = Sr[k] * inv;     // read (f32) precedes aliased write (bf16)
            Pr[k] = (bf16)pv;
        }
    }
    __syncthreads();

    // O = P @ V  : 4 waves each own one (fm, fn) 16x16 output fragment
    const int fm = wave >> 1, fn = wave & 1;
    v8f o = v8f_zero();
    for (int k0 = 0; k0 < Nw; k0 += 32) {
        FragAB pa, vb;
        const bf16* prow = (const bf16*)(S + (size_t)(fm * 16 + lq) * srow) + k0;
        pa.q[0] = *(const uint4*)(prow + lh * 8);
        pa.q[1] = *(const uint4*)(prow + 16 + lh * 8);
        const bf16* vsrc = Vt + (fn * 16 + lq) * vrow + k0;
        vb.q[0] = *(const uint4*)(vsrc + lh * 8);
        vb.q[1] = *(const uint4*)(vsrc + lh * 8 + 16);
        o = wmma_bb(pa.v, vb.v, o);
    }
#pragma unroll
    for (int r = 0; r < 8; ++r) {
        int lrow = qt * 32 + fm * 16 + r + 8 * lh;
        long t = tokrow(lrow);
        p.att[t * 256 + p.cq0 + head * 32 + fn * 16 + lq] = (bf16)o[r];
    }
}

// ---------------------------------------------------------------------------
// LePE: per-window depthwise 3x3 conv on V, added into attention output.
// ---------------------------------------------------------------------------
__global__ __launch_bounds__(256)
void lepe_add(const bf16* __restrict__ qkv, const float* __restrict__ getv_w,
              const float* __restrict__ getv_b, bf16* __restrict__ att) {
    long idx = (long)blockIdx.x * 256 + threadIdx.x;   // over B*L*C
    int c = (int)(idx & 255);
    long tok = idx >> 8;
    int b = (int)(tok >> 10);
    int pos = (int)(tok & 1023);
    int h = pos >> 5, w = pos & 31;
    int br = c >> 6, cc = c & 63;
    int hs = (br & 1) ? 32 : 8;     // WINS: (8,8) (32,8) (8,32) (32,32)
    int wsz = (br & 2) ? 32 : 8;
    int wh = h % hs, ww = w % wsz;
    float acc = getv_b[br * 64 + cc];
#pragma unroll
    for (int dy = 0; dy < 3; ++dy) {
#pragma unroll
        for (int dx = 0; dx < 3; ++dx) {
            int yy = wh + dy - 1, xx = ww + dx - 1;
            if (yy >= 0 && yy < hs && xx >= 0 && xx < wsz) {
                long src = ((long)b * 1024 + (h + dy - 1) * 32 + (w + dx - 1)) * 768
                           + 512 + c;
                acc += (float)qkv[src] * getv_w[((long)(br * 64 + cc)) * 9 + dy * 3 + dx];
            }
        }
    }
    att[tok * 256 + c] = (bf16)((float)att[tok * 256 + c] + acc);
}

// ---------------------------------------------------------------------------
// SGFN depthwise 3x3 (full image, pad 1):  y2f = dwconv(x2n) + bias  (f32)
// ---------------------------------------------------------------------------
__global__ __launch_bounds__(256)
void dw3x3(const bf16* __restrict__ x2n, const float* __restrict__ cw,
           const float* __restrict__ cb, float* __restrict__ y2f) {
    long idx = (long)blockIdx.x * 256 + threadIdx.x;   // over B*L*512
    int c = (int)(idx & 511);
    long tok = idx >> 9;
    int b = (int)(tok >> 10);
    int pos = (int)(tok & 1023);
    int h = pos >> 5, w = pos & 31;
    float acc = cb[c];
#pragma unroll
    for (int dy = 0; dy < 3; ++dy) {
#pragma unroll
        for (int dx = 0; dx < 3; ++dx) {
            int yy = h + dy - 1, xx = w + dx - 1;
            if (yy >= 0 && yy < 32 && xx >= 0 && xx < 32) {
                long src = ((long)b * 1024 + yy * 32 + xx) * 512 + c;
                acc += (float)x2n[src] * cw[(long)c * 9 + dy * 3 + dx];
            }
        }
    }
    y2f[idx] = acc;
}

// gated = bf16((y1f + sg_dil_b) * y2f)
__global__ __launch_bounds__(256)
void gate_kernel(const float* __restrict__ y1f, const float* __restrict__ y2f,
                 const float* __restrict__ bias, bf16* __restrict__ gated) {
    long idx = (long)blockIdx.x * 256 + threadIdx.x;
    int c = (int)(idx & 511);
    gated[idx] = (bf16)((y1f[idx] + bias[c]) * y2f[idx]);
}

// ---------------------------------------------------------------------------
// Host orchestration
// ---------------------------------------------------------------------------
extern "C" void kernel_launch(void* const* d_in, const int* in_sizes, int n_in,
                              void* d_out, int out_size, void* d_ws, size_t ws_size,
                              hipStream_t stream) {
    (void)in_sizes; (void)n_in; (void)out_size; (void)ws_size;
    const float* x      = (const float*)d_in[0];
    const float* g1     = (const float*)d_in[1];
    const float* b1     = (const float*)d_in[2];
    const float* Wqkv   = (const float*)d_in[3];
    const float* bqkv   = (const float*)d_in[4];
    const float* getv_w = (const float*)d_in[5];
    const float* getv_b = (const float*)d_in[6];
    const float* Wproj  = (const float*)d_in[7];
    const float* bproj  = (const float*)d_in[8];
    const float* g2     = (const float*)d_in[9];
    const float* b2     = (const float*)d_in[10];
    const float* Wfc1   = (const float*)d_in[11];
    const float* bfc1   = (const float*)d_in[12];
    const float* gsg    = (const float*)d_in[13];
    const float* bsg    = (const float*)d_in[14];
    const float* sgcw   = (const float*)d_in[15];
    const float* sgcb   = (const float*)d_in[16];
    const float* sgdw   = (const float*)d_in[17];
    const float* sgdb   = (const float*)d_in[18];
    const float* Wfc2   = (const float*)d_in[19];
    const float* bfc2   = (const float*)d_in[20];
    float* out = (float*)d_out;
    char* ws = (char*)d_ws;

    // Workspace arena (bytes). Regions reused once their producer/consumer
    // sequencing allows (all launches are on one stream, in order).
    const size_t OFF_H      = 0;                 // 16 MB  bf16 (ln1 out; later h2, x1n, gated)
    const size_t OFF_QKV    = 16777216;          // 48 MB  bf16 (later x2n @32MB)
    const size_t OFF_X2N    = 33554432;          // inside old qkv region
    const size_t OFF_ATT    = 67108864;          // 16 MB  bf16
    const size_t OFF_XATTN  = 83886080;          // 32 MB  f32
    const size_t OFF_HFEAT  = 117440512;         // 64 MB  bf16 (later y1f f32)
    const size_t OFF_Y1F    = 117440512;
    const size_t OFF_Y2F    = 184549376;         // 64 MB  f32
    const size_t OFF_WQKVT  = 251658240;
    const size_t OFF_WPROJT = OFF_WQKVT  + (size_t)768 * 256 * 2;
    const size_t OFF_WFC1T  = OFF_WPROJT + (size_t)256 * 256 * 2;
    const size_t OFF_WFC2T  = OFF_WFC1T  + (size_t)1024 * 256 * 2;
    const size_t OFF_WDILT  = OFF_WFC2T  + (size_t)256 * 512 * 2;

    bf16* hbuf   = (bf16*)(ws + OFF_H);
    bf16* qkvbuf = (bf16*)(ws + OFF_QKV);
    bf16* attbuf = (bf16*)(ws + OFF_ATT);
    float* xattn = (float*)(ws + OFF_XATTN);
    bf16* h2     = (bf16*)(ws + OFF_H);       // reuse
    bf16* hfeat  = (bf16*)(ws + OFF_HFEAT);
    bf16* x1n    = (bf16*)(ws + OFF_H);       // reuse
    bf16* x2n    = (bf16*)(ws + OFF_X2N);     // reuse
    float* y2f   = (float*)(ws + OFF_Y2F);
    float* y1f   = (float*)(ws + OFF_Y1F);    // reuse (hfeat dead)
    bf16* gated  = (bf16*)(ws + OFF_H);       // reuse (x1n dead after dil gemms)
    bf16* WqkvT  = (bf16*)(ws + OFF_WQKVT);
    bf16* WprojT = (bf16*)(ws + OFF_WPROJT);
    bf16* Wfc1T  = (bf16*)(ws + OFF_WFC1T);
    bf16* Wfc2T  = (bf16*)(ws + OFF_WFC2T);
    bf16* WdilT  = (bf16*)(ws + OFF_WDILT);

    // --- weight transposes / bf16 conversion ---
    transpose_to_bt<<<(256 * 768 + 255) / 256, 256, 0, stream>>>(Wqkv, WqkvT, 256, 768);
    transpose_to_bt<<<(256 * 256 + 255) / 256, 256, 0, stream>>>(Wproj, WprojT, 256, 256);
    transpose_to_bt<<<(256 * 1024 + 255) / 256, 256, 0, stream>>>(Wfc1, Wfc1T, 256, 1024);
    transpose_to_bt<<<(512 * 256 + 255) / 256, 256, 0, stream>>>(Wfc2, Wfc2T, 512, 256);
    dilw_to_bt<<<(9 * 512 * 512 + 255) / 256, 256, 0, stream>>>(sgdw, WdilT);

    // --- LN1 -> h (bf16) ---
    ln_kernel<<<32768, 256, 0, stream>>>(x, nullptr, g1, b1, hbuf, 256, 256, 0);

    // --- qkv = h @ Wqkv + bqkv   (M=32768, N=768, K=256) ---
    GemmParams gp{};
    gp.A = hbuf; gp.Bt = WqkvT; gp.Cb = qkvbuf; gp.bias = bqkv;
    gp.M = 32768; gp.N = 768; gp.K = 256; gp.mode = 0;
    gemm_wmma<<<dim3(6, 256), 256, 0, stream>>>(gp);

    // --- cross-shaped window attention (4 branches) ---
    const int hs[4]  = {8, 32, 8, 32};
    const int wsp[4] = {8, 8, 32, 32};
    for (int br = 0; br < 4; ++br) {
        AttnParams ap{};
        ap.qkv = qkvbuf; ap.att = attbuf; ap.cq0 = br * 64;
        ap.Hsp = hs[br]; ap.Wsp = wsp[br];
        ap.nHw = 32 / hs[br]; ap.nWw = 32 / wsp[br];
        int Nw = hs[br] * wsp[br];
        size_t shmem = (size_t)32 * (Nw + 8) * 4 + (size_t)32 * (Nw + 8) * 2;
        dim3 grid(Nw / 32, 32 * ap.nHw * ap.nWw, 2);
        attn_kernel<<<grid, 128, shmem, stream>>>(ap);
    }
    // LePE adds into attbuf
    lepe_add<<<32768, 256, 0, stream>>>(qkvbuf, getv_w, getv_b, attbuf);

    // --- xattn = x + att @ Wproj + bproj  (f32) ---
    gp = GemmParams{};
    gp.A = attbuf; gp.Bt = WprojT; gp.Cf = xattn; gp.bias = bproj; gp.resid = x;
    gp.M = 32768; gp.N = 256; gp.K = 256; gp.mode = 2;
    gemm_wmma<<<dim3(2, 256), 256, 0, stream>>>(gp);

    // --- LN2 -> h2 ---
    ln_kernel<<<32768, 256, 0, stream>>>(xattn, nullptr, g2, b2, h2, 256, 256, 0);

    // --- hfeat = gelu(h2 @ Wfc1 + bfc1)  (bf16, N=1024) ---
    gp = GemmParams{};
    gp.A = h2; gp.Bt = Wfc1T; gp.Cb = hfeat; gp.bias = bfc1;
    gp.M = 32768; gp.N = 1024; gp.K = 256; gp.mode = 1;
    gemm_wmma<<<dim3(8, 256), 256, 0, stream>>>(gp);

    // --- split LN over 512 channels each ---
    ln_kernel<<<32768, 256, 0, stream>>>(nullptr, hfeat, gsg, bsg, x1n, 512, 1024, 0);
    ln_kernel<<<32768, 256, 0, stream>>>(nullptr, hfeat, gsg, bsg, x2n, 512, 1024, 512);

    // --- y2 = dwconv3x3(x2) + bias (f32) ---
    dw3x3<<<65536, 256, 0, stream>>>(x2n, sgcw, sgcb, y2f);

    // --- y1 = dilated 3x3 conv (dil=2) as 9 shifted implicit-GEMM passes ---
    for (int t = 0; t < 9; ++t) {
        gp = GemmParams{};
        gp.A = x1n; gp.Bt = WdilT + (size_t)t * 512 * 512; gp.Cf = y1f;
        gp.M = 32768; gp.N = 512; gp.K = 512; gp.mode = 3; gp.acc_load = (t > 0);
        gp.dh = (t / 3 - 1) * 2; gp.dw = (t % 3 - 1) * 2; gp.Hd = 32; gp.Wd = 32;
        gemm_wmma<<<dim3(4, 256), 256, 0, stream>>>(gp);
    }

    // --- gate: gated = (y1 + bias) * y2 ---
    gate_kernel<<<65536, 256, 0, stream>>>(y1f, y2f, sgdb, gated);

    // --- out = xattn + gated @ Wfc2 + bfc2 ---
    gp = GemmParams{};
    gp.A = gated; gp.Bt = Wfc2T; gp.Cf = out; gp.bias = bfc2; gp.resid = xattn;
    gp.M = 32768; gp.N = 256; gp.K = 512; gp.mode = 2;
    gemm_wmma<<<dim3(2, 256), 256, 0, stream>>>(gp);
}
